// BiQRNN3DLayer_40243843563920
// MI455X (gfx1250) — matrix operand order repaired and checked by reading, not compiled, not run on hardware
//
#include <hip/hip_runtime.h>
#include <math.h>

typedef _Float16 half8 __attribute__((ext_vector_type(8)));
typedef _Float16 v16h  __attribute__((ext_vector_type(16)));
typedef float    v8f   __attribute__((ext_vector_type(8)));

#define C_IN  16
#define C_OUT 48
#define DDIM  31
#define HDIM  256
#define WDIM  256
#define NG    (C_IN * DDIM * HDIM * WDIM)   // 32,505,856 elements per gate buffer

// Fast device activations: native v_exp_f32 + v_rcp_f32 (TRANS pipe).
__device__ __forceinline__ float fast_sigmoid(float x) {
    return __builtin_amdgcn_rcpf(1.0f + __expf(-x));
}
__device__ __forceinline__ float fast_tanh(float x) {
    // tanh(x) = 1 - 2/(exp(2x)+1)
    return 1.0f - 2.0f * __builtin_amdgcn_rcpf(__expf(2.0f * x) + 1.0f);
}

// ---------------------------------------------------------------------------
// Kernel 0: pack conv weights f32 -> f16 in layout [o=27][n=48][ci=16]
// w (OIDHW): [n=48][ci=16][kd][kh][kw], o = (kd*3+kh)*3+kw
// ---------------------------------------------------------------------------
__global__ void pack_weights_kernel(const float* __restrict__ w,
                                    _Float16* __restrict__ wpack) {
    int i = blockIdx.x * blockDim.x + threadIdx.x;
    if (i >= 27 * 48 * 16) return;
    int ci = i & 15;
    int t  = i >> 4;
    int n  = t % 48;
    int o  = t / 48;
    wpack[(o * 48 + n) * 16 + ci] = (_Float16)w[(n * 16 + ci) * 27 + o];
}

// ---------------------------------------------------------------------------
// Kernel 1: conv3d 16->48 (3x3x3, SAME) + bias + tanh/sigmoid, via WMMA.
// The conv is 27 shifted 1x1 convs over 16 channels; we pair two kernel
// offsets per v_wmma_f32_16x16x32_f16 so all K=32 lanes are real data:
//   K = 16*pair + ci, pair p -> kernel offset o = 2c + p.
//
// Fragment layouts (wave32, CDNA5 ISA 7.12.2):
//  A 16x32 f16: lane l (M = l%16): elems 0-15 = K {16*hi + 0..15}, ci = elem,
//               of offset o0 (hi=0) or o1 (hi=1) (hi = l>=16).
//               => two contiguous 16B LDS reads + concat shuffle.
//  B 32x16 f16: lanes 0-15 hold K 0-15 (offset o0), lanes 16-31 hold K 16-31
//               (offset o1), N = l%16, elem e = ci.
//               => one contiguous 32B LDS read per lane from slot o = 2c+hi.
//  C/D 16x16 f32: VGPR r -> M = r + 8*hi, N = l%16.
// 27 is odd: weight slot 27 in LDS is zero-filled; the tail chunk uses it for
// B lanes 16-31; A lanes 16-31 still hold offset 26.
// ---------------------------------------------------------------------------
__global__ __launch_bounds__(128)
void conv_gates_kernel(const float* __restrict__ x,
                       const _Float16* __restrict__ wpack,
                       const float* __restrict__ bias,
                       float* __restrict__ Z,
                       float* __restrict__ F1,
                       float* __restrict__ F2) {
    __shared__ __align__(16) _Float16 lds_x[3 * 3 * 66 * 16]; // [dz][hy][wx][ci]
    __shared__ __align__(16) _Float16 lds_w[28 * 48 * 16];    // [o][n][ci], o=27 zeros

    const int tid = threadIdx.x;
    const int w0  = blockIdx.x * 64;
    const int h   = blockIdx.y;
    const int d   = blockIdx.z;

    // ---- stage packed weights into LDS (b128 copies, L2-resident source) ----
    {
        const uint4* src = (const uint4*)wpack;
        uint4*       dst = (uint4*)lds_w;
        for (int i = tid; i < (27 * 48 * 16) / 8; i += 128) dst[i] = src[i];
        const uint4 z4 = make_uint4(0u, 0u, 0u, 0u);
        for (int i = tid; i < (48 * 16) / 8; i += 128)
            dst[(27 * 48 * 16) / 8 + i] = z4;              // zero slot o=27
    }

    // ---- stage x halo tile (16ch x 3 x 3 x 66) f32 -> f16, coalesced in w ----
    for (int i = tid; i < 3 * 3 * 66 * 16; i += 128) {
        int wx = i % 66;
        int j  = i / 66;
        int hy = j % 3;
        int k  = j / 3;
        int dz = k % 3;
        int ci = k / 3;
        int dd = d + dz - 1, hh = h + hy - 1, ww = w0 + wx - 1;
        float v = 0.0f;
        if ((unsigned)dd < DDIM && (unsigned)hh < HDIM && (unsigned)ww < WDIM)
            v = x[((ci * DDIM + dd) * HDIM + hh) * WDIM + ww];
        lds_x[((dz * 3 + hy) * 66 + wx) * 16 + ci] = (_Float16)v;
    }
    __syncthreads();

    const int lane = tid & 31;
    const int wv   = tid >> 5;     // wave id 0..3 -> 16-position sub-tile
    const int m    = lane & 15;    // A: row M ; B: col N (channel within group)
    const int hi   = lane >> 4;    // half-wave select
    const int xrow = wv * 16 + m;  // output position within the 64-wide tile

    v8f acc0 = {}, acc1 = {}, acc2 = {};
    const half8 hzero = {};

    #pragma unroll
    for (int c = 0; c < 13; ++c) {
        const int o0 = 2 * c, o1 = 2 * c + 1;
        const int kd0 = o0 / 9, kh0 = (o0 / 3) % 3, kw0 = o0 % 3;
        const int kd1 = o1 / 9, kh1 = (o1 / 3) % 3, kw1 = o1 % 3;

        const int ka = hi ? ((kd1 * 3 + kh1) * 66) + kw1 : ((kd0 * 3 + kh0) * 66) + kw0;
        const half8 alo = *(const half8*)
            &lds_x[(ka + xrow) * 16];
        const half8 ahi = *(const half8*)
            &lds_x[(ka + xrow) * 16 + 8];
        const v16h a = __builtin_shufflevector(alo, ahi,
                          0,1,2,3,4,5,6,7,8,9,10,11,12,13,14,15);

        const int ob = (2 * c + hi) * 48;   // per-lane weight slot (o0 / o1)
        const v16h b0 = *(const v16h*)&lds_w[(ob +  0 + m) * 16];
        const v16h b1 = *(const v16h*)&lds_w[(ob + 16 + m) * 16];
        const v16h b2 = *(const v16h*)&lds_w[(ob + 32 + m) * 16];

        acc0 = __builtin_amdgcn_wmma_f32_16x16x32_f16(false, a, false, b0,
                                                      (short)0, acc0, false, false);
        acc1 = __builtin_amdgcn_wmma_f32_16x16x32_f16(false, a, false, b1,
                                                      (short)0, acc1, false, false);
        acc2 = __builtin_amdgcn_wmma_f32_16x16x32_f16(false, a, false, b2,
                                                      (short)0, acc2, false, false);
    }

    // ---- tail chunk: offset o=26 real, pair slot o=27 is zeros ----
    {
        const half8 alo = *(const half8*)
            &lds_x[(((2 * 3 + 2) * 66) + xrow + 2) * 16];
        const half8 ahi = *(const half8*)
            &lds_x[(((2 * 3 + 2) * 66) + xrow + 2) * 16 + 8];
        const v16h a = __builtin_shufflevector(alo, ahi,
                          0,1,2,3,4,5,6,7,8,9,10,11,12,13,14,15);

        const int ob = (26 + hi) * 48;      // hi lanes read the zero slot
        const v16h b0 = *(const v16h*)&lds_w[(ob +  0 + m) * 16];
        const v16h b1 = *(const v16h*)&lds_w[(ob + 16 + m) * 16];
        const v16h b2 = *(const v16h*)&lds_w[(ob + 32 + m) * 16];

        acc0 = __builtin_amdgcn_wmma_f32_16x16x32_f16(false, a, false, b0,
                                                      (short)0, acc0, false, false);
        acc1 = __builtin_amdgcn_wmma_f32_16x16x32_f16(false, a, false, b1,
                                                      (short)0, acc1, false, false);
        acc2 = __builtin_amdgcn_wmma_f32_16x16x32_f16(false, a, false, b2,
                                                      (short)0, acc2, false, false);
    }

    // ---- fused bias + activations (native exp/rcp), vector gate stores ----
    const float bz  = bias[m];
    const float bf1 = bias[16 + m];
    const float bf2 = bias[32 + m];

    float zo[8], f1o[8], f2o[8];
    #pragma unroll
    for (int r = 0; r < 8; ++r) {
        zo[r]  = fast_tanh(acc0[r] + bz);
        f1o[r] = fast_sigmoid(acc1[r] + bf1);
        f2o[r] = fast_sigmoid(acc2[r] + bf2);
    }

    // D-fragment: channel = m, position = w0 + wv*16 + 8*hi + r
    const int wpos = w0 + wv * 16 + 8 * hi;
    const int gidx = ((m * DDIM + d) * HDIM + h) * WDIM + wpos;

    *(float4*)&Z[gidx]      = make_float4(zo[0],  zo[1],  zo[2],  zo[3]);
    *(float4*)&Z[gidx + 4]  = make_float4(zo[4],  zo[5],  zo[6],  zo[7]);
    *(float4*)&F1[gidx]     = make_float4(f1o[0], f1o[1], f1o[2], f1o[3]);
    *(float4*)&F1[gidx + 4] = make_float4(f1o[4], f1o[5], f1o[6], f1o[7]);
    *(float4*)&F2[gidx]     = make_float4(f2o[0], f2o[1], f2o[2], f2o[3]);
    *(float4*)&F2[gidx + 4] = make_float4(f2o[4], f2o[5], f2o[6], f2o[7]);
}

// ---------------------------------------------------------------------------
// Kernel 2: bidirectional scan over depth (D=31), h = f*h + (1-f)*z.
// One lane per (c,h,w); forward partials kept in 31 registers so F1 and F2
// are read once, Z twice, and the output written exactly once.
// ---------------------------------------------------------------------------
__global__ __launch_bounds__(256)
void scan_kernel(const float* __restrict__ Z,
                 const float* __restrict__ F1,
                 const float* __restrict__ F2,
                 float* __restrict__ out) {
    const int g   = blockIdx.x * 256 + threadIdx.x;     // < 16*256*256
    const int c   = g >> 16;
    const int rem = g & 65535;                          // h*256 + w
    const int sd  = HDIM * WDIM;                        // 65536
    const int base = c * DDIM * sd + rem;

    float hl[DDIM];
    float hacc = 0.0f;
    #pragma unroll
    for (int d = 0; d < DDIM; ++d) {
        const int idx = base + d * sd;
        const float z = Z[idx];
        const float f = F1[idx];
        hacc = f * hacc + (1.0f - f) * z;
        hl[d] = hacc;
    }

    hacc = 0.0f;
    #pragma unroll
    for (int d = DDIM - 1; d >= 0; --d) {
        const int idx = base + d * sd;
        const float z = Z[idx];
        const float f = F2[idx];
        hacc = f * hacc + (1.0f - f) * z;
        out[idx] = hl[d] + hacc;
    }
}

// ---------------------------------------------------------------------------
// Launcher. Workspace layout: [wpack f16, padded to 64KB][Z][F1][F2] (f32).
// Requires ws_size >= 65536 + 3*NG*4  (~390 MB).
// ---------------------------------------------------------------------------
extern "C" void kernel_launch(void* const* d_in, const int* in_sizes, int n_in,
                              void* d_out, int out_size, void* d_ws, size_t ws_size,
                              hipStream_t stream) {
    const float* x    = (const float*)d_in[0];
    const float* w    = (const float*)d_in[1];
    const float* bias = (const float*)d_in[2];
    float*       out  = (float*)d_out;

    _Float16* wpack = (_Float16*)d_ws;
    float*    Z     = (float*)((char*)d_ws + 65536);
    float*    F1    = Z + (size_t)NG;
    float*    F2    = Z + 2 * (size_t)NG;

    pack_weights_kernel<<<(27 * 48 * 16 + 255) / 256, 256, 0, stream>>>(w, wpack);

    dim3 grid1(WDIM / 64, HDIM, DDIM);   // (4, 256, 31)
    conv_gates_kernel<<<grid1, 128, 0, stream>>>(x, wpack, bias, Z, F1, F2);

    scan_kernel<<<(C_IN * HDIM * WDIM) / 256, 256, 0, stream>>>(Z, F1, F2, out);
}
